// OTClassifier_36825049596631
// MI455X (gfx1250) — matrix operand order repaired and checked by reading, not compiled, not run on hardware
//
#include <hip/hip_runtime.h>
#include <math.h>

#define KEPS 0.0025f   // blur^p = 0.05^2

typedef __attribute__((ext_vector_type(16))) _Float16 v16h;
typedef __attribute__((ext_vector_type(8)))  _Float16 v8h;
typedef __attribute__((ext_vector_type(8)))  float    v8f;

// ---------------------------------------------------------------------------
// Fragment loaders: contiguous K order (not the ISA table order), same for A and B
// A 16x32 f16: lane holds row M=lane&15; slots 0..15 = K=kb..kb+15,
//              kb = (lane>=16)?16:0.
// B 32x16 f16: lane holds col N=lane&15; slots 0..15 = K=kb..kb+15,
//              kb = (lane>=16)?16:0  (16 contiguous halves -> 2x b128 loads).
// ---------------------------------------------------------------------------
__device__ __forceinline__ v16h load_a_frag(const _Float16* p) {
  v8h lo = *(const v8h*)p;          // K = kb .. kb+7
  v8h hg = *(const v8h*)(p + 8);    // K = kb+8 .. kb+15
  v16h a;
#pragma unroll
  for (int i = 0; i < 8; ++i) { a[i] = lo[i]; a[8 + i] = hg[i]; }
  return a;
}

// ---------------------------------------------------------------------------
// Elementwise helpers
// ---------------------------------------------------------------------------
__global__ void cvt_f16_kernel(const float* __restrict__ in, _Float16* __restrict__ out, int n) {
  int i = blockIdx.x * blockDim.x + threadIdx.x;
  if (i < n) out[i] = (_Float16)in[i];
}

// W (K x N, row-major f32) -> WT (N x K, row-major f16)
__global__ void cvt_t_f16_kernel(const float* __restrict__ W, _Float16* __restrict__ WT,
                                 int K, int N) {
  int i = blockIdx.x * blockDim.x + threadIdx.x;
  if (i >= K * N) return;
  int k = i / N, n = i % N;
  WT[(size_t)n * K + k] = (_Float16)W[i];
}

__global__ void fill_f32_kernel(float* __restrict__ p, float v, int n) {
  int i = blockIdx.x * blockDim.x + threadIdx.x;
  if (i < n) p[i] = v;
}

// nrm[r] = 0.5 * sum_k X[r][k]^2
__global__ void rownorm_kernel(const _Float16* __restrict__ X, float* __restrict__ nrm,
                               int rows, int K) {
  int r = blockIdx.x * blockDim.x + threadIdx.x;
  if (r >= rows) return;
  const _Float16* p = X + (size_t)r * K;
  float s = 0.f;
  for (int k = 0; k < K; ++k) { float v = (float)p[k]; s += v * v; }
  nrm[r] = 0.5f * s;
}

// ---------------------------------------------------------------------------
// Generic WMMA GEMM: C[M][Nc] = act(A[M][K] @ BT[Nc][K]^T + bias), f16 out.
// One 16x16 tile per wave, 4 waves per block; grid must be exact.
// ---------------------------------------------------------------------------
__global__ void gemm16_kernel(const _Float16* __restrict__ A, const _Float16* __restrict__ BT,
                              const float* __restrict__ bias, _Float16* __restrict__ C,
                              int Mt, int Nt, int K, int relu) {
  const int wv = threadIdx.x >> 5, lane = threadIdx.x & 31;
  const int l15 = lane & 15, hi = lane >> 4;
  const int tile = blockIdx.x * 4 + wv;
  if (tile >= Mt * Nt) return;
  const int rt = tile / Nt, ct = tile % Nt;

  const _Float16* arow = A + (size_t)(rt * 16 + l15) * K + (hi ? 16 : 0);
  const _Float16* brow = BT + (size_t)(ct * 16 + l15) * K + (hi ? 16 : 0);

  v8f c = {};
  for (int kc = 0; kc < K; kc += 32) {
    v16h a = load_a_frag(arow + kc);
    v16h b = *(const v16h*)(brow + kc);
    c = __builtin_amdgcn_wmma_f32_16x16x32_f16(false, a, false, b, (short)0, c, false, false);
  }
  const int col = ct * 16 + l15;
  const float bv = bias[col];
  const int Nc = Nt * 16;
#pragma unroll
  for (int r = 0; r < 8; ++r) {
    float v = c[r] + bv;
    if (relu) v = fmaxf(v, 0.f);
    C[(size_t)(rt * 16 + r + 8 * hi) * Nc + col] = (_Float16)v;
  }
}

// ---------------------------------------------------------------------------
// Fused Gram + online log-sum-exp: symmetric Sinkhorn iteration on OT(x,x).
// t_i = nx_i - eps*LSE_j(loga + (p_j - C_ij)/eps), C_ij = max(nx_i+nx_j-xp_i.xp_j,0)
// mode 0: p_out = 0.5*(p_in + t);  mode 1: p_out = t  (final extrapolation)
// 128 threads (4 waves), wave w owns rows [blockIdx*64 + 16w, +16). Grid=N/64.
// ---------------------------------------------------------------------------
__global__ void sym_x_kernel(const _Float16* __restrict__ xp, const float* __restrict__ nx,
                             const float* __restrict__ p_in, float* __restrict__ p_out,
                             int N, float loga, int mode) {
  const int wv = threadIdx.x >> 5, lane = threadIdx.x & 31;
  const int l15 = lane & 15, hi = lane >> 4;
  const int rt = blockIdx.x * 4 + wv;
  const int row0 = rt * 16;

  // cache A fragments for our 16 rows (K = 128 -> 4 chunks of 32)
  v16h afrag[4];
  {
    const _Float16* arow = xp + (size_t)(row0 + l15) * 128 + (hi ? 16 : 0);
#pragma unroll
    for (int cch = 0; cch < 4; ++cch) afrag[cch] = load_a_frag(arow + cch * 32);
  }
  float nxr[8];
#pragma unroll
  for (int r = 0; r < 8; ++r) nxr[r] = nx[row0 + r + 8 * hi];

  float mx[8], sm[8];
#pragma unroll
  for (int r = 0; r < 8; ++r) { mx[r] = -3.0e38f; sm[r] = 0.f; }

  const float invEps = 1.0f / KEPS;
  const int jTiles = N >> 4;
  const int kbB = hi ? 16 : 0;
  for (int jt = 0; jt < jTiles; ++jt) {
    const int jcol = jt * 16 + l15;
    const float nxj = nx[jcol];
    const float pj = p_in[jcol];
    const _Float16* brow = xp + (size_t)jcol * 128 + kbB;
    v8f c = {};
#pragma unroll
    for (int cch = 0; cch < 4; ++cch) {
      v16h b = *(const v16h*)(brow + cch * 32);
      c = __builtin_amdgcn_wmma_f32_16x16x32_f16(false, afrag[cch], false, b, (short)0, c,
                                                 false, false);
    }
#pragma unroll
    for (int r = 0; r < 8; ++r) {
      float Cv = fmaxf(nxr[r] + nxj - c[r], 0.f);
      float a = loga + (pj - Cv) * invEps;
      float mn = fmaxf(mx[r], a);
      sm[r] = sm[r] * __expf(mx[r] - mn) + __expf(a - mn);
      mx[r] = mn;
    }
  }
  // merge running (max,sum) across the 16 lanes holding each row
#pragma unroll
  for (int r = 0; r < 8; ++r) {
    float m = mx[r], s = sm[r];
#pragma unroll
    for (int d = 1; d < 16; d <<= 1) {
      float mo = __shfl_xor(m, d, 32);
      float so = __shfl_xor(s, d, 32);
      float mn = fmaxf(m, mo);
      s = s * __expf(m - mn) + so * __expf(mo - mn);
      m = mn;
    }
    if (l15 == 0) {
      int row = row0 + r + 8 * hi;
      float t = nxr[r] - KEPS * (m + __logf(s));
      p_out[row] = mode ? t : 0.5f * (p_in[row] + t);
    }
  }
}

// ---------------------------------------------------------------------------
// CxyT[class][j][i] = max(nx_i + np[class*16+j] - xp_i . proto[class][j], 0)
// WMMA tile per wave; B = prototypes (rows contiguous in K). Grid exact.
// ---------------------------------------------------------------------------
__global__ void cxy_kernel(const _Float16* __restrict__ xp, const _Float16* __restrict__ pr,
                           const float* __restrict__ nx, const float* __restrict__ np,
                           float* __restrict__ CxyT, int Nt, int N) {
  const int wv = threadIdx.x >> 5, lane = threadIdx.x & 31;
  const int l15 = lane & 15, hi = lane >> 4;
  const int tile = blockIdx.x * 4 + wv;
  const int cls = tile / Nt, rt = tile % Nt;

  const _Float16* arow = xp + (size_t)(rt * 16 + l15) * 128 + (hi ? 16 : 0);
  const _Float16* brow = pr + ((size_t)cls * 16 + l15) * 128 + (hi ? 16 : 0);
  v8f c = {};
#pragma unroll
  for (int cch = 0; cch < 4; ++cch) {
    v16h a = load_a_frag(arow + cch * 32);
    v16h b = *(const v16h*)(brow + cch * 32);
    c = __builtin_amdgcn_wmma_f32_16x16x32_f16(false, a, false, b, (short)0, c, false, false);
  }
  const int j = l15;
  const float npj = np[cls * 16 + j];
#pragma unroll
  for (int r = 0; r < 8; ++r) {
    int i = rt * 16 + r + 8 * hi;
    float Cv = fmaxf(nx[i] + npj - c[r], 0.f);
    CxyT[((size_t)cls * 16 + j) * N + i] = Cv;
  }
}

// g[cj] = -eps * LSE_i(loga + (f[class][i] - CxyT[cj][i])/eps). Grid = 1600.
__global__ void g_update_kernel(const float* __restrict__ CxyT, const float* __restrict__ f,
                                float* __restrict__ g, int N, float loga) {
  __shared__ float smx[256], ssm[256];
  const int cj = blockIdx.x, cls = cj >> 4, t = threadIdx.x;
  const float* Crow = CxyT + (size_t)cj * N;
  const float* fr = f + (size_t)cls * N;
  const float invEps = 1.f / KEPS;
  float m = -3.0e38f, s = 0.f;
  for (int i = t; i < N; i += 256) {
    float a = loga + (fr[i] - Crow[i]) * invEps;
    float mn = fmaxf(m, a);
    s = s * __expf(m - mn) + __expf(a - mn);
    m = mn;
  }
  smx[t] = m; ssm[t] = s;
  __syncthreads();
  for (int d = 128; d > 0; d >>= 1) {
    if (t < d) {
      float mo = smx[t + d], so = ssm[t + d];
      float mn = fmaxf(smx[t], mo);
      ssm[t] = ssm[t] * __expf(smx[t] - mn) + so * __expf(mo - mn);
      smx[t] = mn;
    }
    __syncthreads();
  }
  if (t == 0) g[cj] = -KEPS * (smx[0] + __logf(ssm[0]));
}

// f[class][i] = -eps * LSE_j(logb + (g[class][j] - CxyT[class][j][i])/eps)
// Grid = 100*bpc blocks, 256 threads; bpc = N/256.
__global__ void f_update_kernel(const float* __restrict__ CxyT, const float* __restrict__ g,
                                float* __restrict__ f, int N, int bpc, float logb) {
  __shared__ float gs[16];
  const int cls = blockIdx.x / bpc;
  const int i = (blockIdx.x % bpc) * 256 + threadIdx.x;
  if (threadIdx.x < 16) gs[threadIdx.x] = g[cls * 16 + threadIdx.x];
  __syncthreads();
  const float invEps = 1.f / KEPS;
  const float* Cb = CxyT + (size_t)cls * 16 * N + i;
  float a[16], m = -3.0e38f;
#pragma unroll
  for (int j = 0; j < 16; ++j) {
    a[j] = logb + (gs[j] - Cb[(size_t)j * N]) * invEps;
    m = fmaxf(m, a[j]);
  }
  float s = 0.f;
#pragma unroll
  for (int j = 0; j < 16; ++j) s += __expf(a[j] - m);
  f[(size_t)cls * N + i] = -KEPS * (m + __logf(s));
}

// Per-class 16x16 self-term OT(a,a): all in LDS. Grid = 100 blocks x 256.
__global__ void cyy_sym_kernel(const float* __restrict__ proto, float* __restrict__ py,
                               float logb) {
  __shared__ float Cyy[256];
  __shared__ float nrm[16];
  __shared__ float p[16];
  const int cls = blockIdx.x, t = threadIdx.x;
  const float* P = proto + (size_t)cls * 16 * 128;
  if (t < 16) {
    float s = 0.f;
    for (int k = 0; k < 128; ++k) { float v = P[t * 128 + k]; s += v * v; }
    nrm[t] = 0.5f * s; p[t] = 0.f;
  }
  __syncthreads();
  {
    int j1 = t >> 4, j2 = t & 15;
    float d = 0.f;
    for (int k = 0; k < 128; ++k) d += P[j1 * 128 + k] * P[j2 * 128 + k];
    Cyy[t] = fmaxf(nrm[j1] + nrm[j2] - d, 0.f);
  }
  __syncthreads();
  const float invEps = 1.f / KEPS;
  for (int it = 0; it < 20; ++it) {
    float tt = 0.f;
    if (t < 16) {
      float m = -3.0e38f, s = 0.f;
      for (int j2 = 0; j2 < 16; ++j2) {
        float a = logb + (p[j2] - Cyy[t * 16 + j2]) * invEps;
        float mn = fmaxf(m, a);
        s = s * __expf(m - mn) + __expf(a - mn);
        m = mn;
      }
      tt = -KEPS * (m + __logf(s));
    }
    __syncthreads();
    if (t < 16) p[t] = 0.5f * (p[t] + tt);
    __syncthreads();
  }
  if (t < 16) {
    float m = -3.0e38f, s = 0.f;
    for (int j2 = 0; j2 < 16; ++j2) {
      float a = logb + (p[j2] - Cyy[t * 16 + j2]) * invEps;
      float mn = fmaxf(m, a);
      s = s * __expf(m - mn) + __expf(a - mn);
      m = mn;
    }
    py[cls * 16 + t] = -KEPS * (m + __logf(s));
  }
}

// out[class] = -( mean_i(f - px) + mean_j(g - py) ). Grid = 100 x 256.
__global__ void result_kernel(const float* __restrict__ f, const float* __restrict__ px,
                              const float* __restrict__ g, const float* __restrict__ py,
                              float* __restrict__ out, int N) {
  __shared__ float red[256];
  const int cls = blockIdx.x, t = threadIdx.x;
  const float* fr = f + (size_t)cls * N;
  float s = 0.f;
  for (int i = t; i < N; i += 256) s += fr[i] - px[i];
  red[t] = s;
  __syncthreads();
  for (int d = 128; d > 0; d >>= 1) {
    if (t < d) red[t] += red[t + d];
    __syncthreads();
  }
  if (t == 0) {
    float sg = 0.f;
    for (int j = 0; j < 16; ++j) sg += g[cls * 16 + j] - py[cls * 16 + j];
    out[cls] = -(red[0] / (float)N + sg / 16.0f);
  }
}

// ---------------------------------------------------------------------------
// Host orchestration
// ---------------------------------------------------------------------------
extern "C" void kernel_launch(void* const* d_in, const int* in_sizes, int n_in,
                              void* d_out, int out_size, void* d_ws, size_t ws_size,
                              hipStream_t stream) {
  const int N = 6144, IN = 512, HID = 256, PROJ = 128, NC = 100, M = 16;
  const float loga = -logf((float)N);
  const float logb = -logf((float)M);

  const float* x  = (const float*)d_in[0];
  const float* W1 = (const float*)d_in[1];
  const float* b1 = (const float*)d_in[2];
  const float* W2 = (const float*)d_in[3];
  const float* b2 = (const float*)d_in[4];
  const float* pf = (const float*)d_in[5];
  float* out = (float*)d_out;

  // workspace carve-out (256B aligned slices)
  char* ws = (char*)d_ws;
  size_t off = 0;
  auto take = [&](size_t bytes) -> char* {
    char* p = ws + off;
    off = (off + bytes + 255) & ~(size_t)255;
    return p;
  };
  _Float16* x_h  = (_Float16*)take((size_t)N * IN * 2);
  _Float16* w1t  = (_Float16*)take((size_t)HID * IN * 2);
  _Float16* w2t  = (_Float16*)take((size_t)PROJ * HID * 2);
  _Float16* h_h  = (_Float16*)take((size_t)N * HID * 2);
  _Float16* xp_h = (_Float16*)take((size_t)N * PROJ * 2);
  _Float16* pr_h = (_Float16*)take((size_t)NC * M * PROJ * 2);
  float* nx  = (float*)take((size_t)N * 4);
  float* np  = (float*)take((size_t)NC * M * 4);
  float* p_a = (float*)take((size_t)N * 4);
  float* p_b = (float*)take((size_t)N * 4);
  float* px  = (float*)take((size_t)N * 4);
  float* CxyT = (float*)take((size_t)NC * M * N * 4);
  float* fbuf = (float*)take((size_t)NC * N * 4);
  float* gbuf = (float*)take((size_t)NC * M * 4);
  float* pybuf = (float*)take((size_t)NC * M * 4);
  (void)ws_size; (void)in_sizes; (void)n_in; (void)out_size;

  // 1) precision conversions
  {
    int n = N * IN;
    cvt_f16_kernel<<<(n + 255) / 256, 256, 0, stream>>>(x, x_h, n);
  }
  {
    int n = NC * M * PROJ;
    cvt_f16_kernel<<<(n + 255) / 256, 256, 0, stream>>>(pf, pr_h, n);
  }
  {
    int n = IN * HID;
    cvt_t_f16_kernel<<<(n + 255) / 256, 256, 0, stream>>>(W1, w1t, IN, HID);
  }
  {
    int n = HID * PROJ;
    cvt_t_f16_kernel<<<(n + 255) / 256, 256, 0, stream>>>(W2, w2t, HID, PROJ);
  }

  // 2) projector GEMMs (WMMA)
  {
    int Mt = N / 16, Nt = HID / 16;                 // 384 x 16 tiles
    gemm16_kernel<<<(Mt * Nt) / 4, 128, 0, stream>>>(x_h, w1t, b1, h_h, Mt, Nt, IN, 1);
  }
  {
    int Mt = N / 16, Nt = PROJ / 16;                // 384 x 8 tiles
    gemm16_kernel<<<(Mt * Nt) / 4, 128, 0, stream>>>(h_h, w2t, b2, xp_h, Mt, Nt, HID, 0);
  }

  // 3) half-squared-norms
  rownorm_kernel<<<(N + 255) / 256, 256, 0, stream>>>(xp_h, nx, N, PROJ);
  rownorm_kernel<<<(NC * M + 255) / 256, 256, 0, stream>>>(pr_h, np, NC * M, PROJ);

  // 4) symmetric Sinkhorn on OT(x,x): fused WMMA-Gram + online LSE
  fill_f32_kernel<<<(N + 255) / 256, 256, 0, stream>>>(p_a, 0.f, N);
  float* pc = p_a;
  float* pn = p_b;
  for (int it = 0; it < 20; ++it) {
    sym_x_kernel<<<N / 64, 128, 0, stream>>>(xp_h, nx, pc, pn, N, loga, 0);
    float* tmp = pc; pc = pn; pn = tmp;
  }
  sym_x_kernel<<<N / 64, 128, 0, stream>>>(xp_h, nx, pc, px, N, loga, 1);

  // 5) cross costs CxyT (WMMA) + cross Sinkhorn iterations
  {
    int Nt = N / 16;                                // 100*384 tiles
    cxy_kernel<<<(NC * Nt) / 4, 128, 0, stream>>>(xp_h, pr_h, nx, np, CxyT, Nt, N);
  }
  fill_f32_kernel<<<(NC * N + 255) / 256, 256, 0, stream>>>(fbuf, 0.f, NC * N);
  const int bpc = N / 256;
  for (int it = 0; it < 20; ++it) {
    g_update_kernel<<<NC * M, 256, 0, stream>>>(CxyT, fbuf, gbuf, N, loga);
    f_update_kernel<<<NC * bpc, 256, 0, stream>>>(CxyT, gbuf, fbuf, N, bpc, logb);
  }
  // final extrapolation: f_out == f (last update already used final g);
  // g_out = one more g-update with final f.
  g_update_kernel<<<NC * M, 256, 0, stream>>>(CxyT, fbuf, gbuf, N, loga);

  // 6) per-class self term OT(y,y) and final assembly
  cyy_sym_kernel<<<NC, 256, 0, stream>>>(pf, pybuf, logb);
  result_kernel<<<NC, 256, 0, stream>>>(fbuf, px, gbuf, pybuf, out, N);
}
